// Siege_60112362274858
// MI455X (gfx1250) — matrix operand order repaired and checked by reading, not compiled, not run on hardware
//
#include <hip/hip_runtime.h>
#include <hip/hip_bf16.h>
#include <math.h>

// ---------------------------------------------------------------------------
// Graph conv net on MI455X (gfx1250, wave32, WMMA).
// g = [x_self | x_nbr | e] @ W  done as f16 WMMA (K=256) + f32 rank-1 edge term.
// Two matmul passes per layer (stats pass + apply pass) to avoid materializing
// the 268MB hidden tensor; everything stays L2-resident.
// ---------------------------------------------------------------------------

typedef _Float16 h16;
typedef __attribute__((ext_vector_type(8)))  _Float16 v8h;
typedef __attribute__((ext_vector_type(16))) _Float16 v16h;
typedef __attribute__((ext_vector_type(8)))  float    v8f;

#define B_SZ   8
#define N_SZ   1024
#define M_SZ   32
#define HA     128
#define HC     256        // 2*HA
#define KD     256        // WMMA K depth (edge column handled separately)
#define NPB    4          // nodes per block
#define NBLK   ((B_SZ*N_SZ)/NPB)   // 2048
#define APAD   264        // LDS A-tile row stride (halves), bank-conflict pad
#define GPAD   260        // LDS g-tile row stride (floats)
#define EPS    1e-5f

// ---------------------------------------------------------------------------
__global__ void embed_kernel(const int* __restrict__ na,
                             const float* __restrict__ emb,
                             float* __restrict__ x) {
  int idx = blockIdx.x * blockDim.x + threadIdx.x;   // B*N*HA threads
  x[idx] = emb[na[idx >> 7] * HA + (idx & 127)];
}

// W[l] row-major (257 x 256) f32  ->  WhT[c][k] f16, k<256 (transposed)
__global__ void convw_kernel(const float* __restrict__ Wl, h16* __restrict__ WhT) {
  int k = threadIdx.x;        // 0..255
  int c = blockIdx.x;         // 0..255
  WhT[c * KD + k] = (h16)Wl[k * HC + c];
}

// ---------------------------------------------------------------------------
// Shared matmul body: fills acc[4] (16x16 f32 tiles) for one node's 32x256 g.
// A fragment layout (16-bit A 16x32): lane<16 -> K kb+0..15,
//                                     lane>=16 -> K kb+16..31.
// B fragment layout (16-bit B 32x16): lane<16 -> K kb+0..15, lane>=16 -> kb+16..31,
//                                     N = lane&15  (WhT[c][k] gives contiguous 32B).
// ---------------------------------------------------------------------------
__device__ __forceinline__ void node_matmul(const h16* ldsA, const h16* __restrict__ WhT,
                                            int wr, int wc, int l15, int half_sel,
                                            v8f acc[4]) {
#pragma unroll
  for (int t = 0; t < 4; ++t)
#pragma unroll
    for (int j = 0; j < 8; ++j) acc[t][j] = 0.f;

#pragma unroll
  for (int kt = 0; kt < 8; ++kt) {
    int kb = kt * 32;
    const h16* ap = ldsA + (wr * 16 + l15) * APAD + kb + 16 * half_sel;
    v8h alo = *(const v8h*)ap;
    v8h ahi = *(const v8h*)(ap + 8);
    v16h a;
#pragma unroll
    for (int j = 0; j < 8; ++j) { a[j] = alo[j]; a[8 + j] = ahi[j]; }
#pragma unroll
    for (int t = 0; t < 4; ++t) {
      int cb = wc * 64 + t * 16;
      v16h bf = *(const v16h*)(WhT + (cb + l15) * KD + kb + 16 * half_sel);
      acc[t] = __builtin_amdgcn_wmma_f32_16x16x32_f16(
          false, a, false, bf, (short)0, acc[t], false, false);
    }
  }
}

__device__ __forceinline__ void fill_tile(const float* __restrict__ x,
                                          const int* __restrict__ eidx,
                                          const float* __restrict__ eattr,
                                          int node, int tid,
                                          h16* ldsA, float* ldsE) {
  int b_i = node >> 10;
  // self embedding: replicate x[node,:] into rows m=0..31, cols 0..127
  {
    h16 xh = (h16)x[node * HA + (tid & 127)];
    for (int m = (tid >> 7); m < 32; m += 2) ldsA[m * APAD + (tid & 127)] = xh;
  }
  // neighbor embeddings: cols 128..255 ; 8 threads per edge row, 16 floats each
  {
    int m  = tid >> 3;
    int cs = (tid & 7) * 16;
    int nb = eidx[node * M_SZ + m];
    const float* src = x + (b_i * N_SZ + nb) * HA + cs;
#pragma unroll
    for (int j = 0; j < 16; ++j) ldsA[m * APAD + HA + cs + j] = (h16)src[j];
  }
  if (tid < M_SZ) ldsE[tid] = eattr[node * M_SZ + tid];
}

// ---------------------------------------------------------------------------
// Pass 1: per-channel sum / sum-of-squares of g over this block's 4*32 rows.
// Deterministic: contributions staged in LDS, channel-owner thread reduces.
// ---------------------------------------------------------------------------
__global__ void __launch_bounds__(256)
stats_kernel(const float* __restrict__ x, const int* __restrict__ eidx,
             const float* __restrict__ eattr, const h16* __restrict__ WhT,
             const float* __restrict__ bias, const float* __restrict__ wlast,
             float* __restrict__ partH) {
  __shared__ h16   ldsA[32 * APAD];
  __shared__ float ldsE[M_SZ];
  __shared__ float st1[4][HC];
  __shared__ float st2[4][HC];

  int tid = threadIdx.x;
  int lane = tid & 31, wid = tid >> 5;
  int wr = wid >> 2, wc = wid & 3;
  int half_sel = lane >> 4, l15 = lane & 15;
  float reg_s1 = 0.f, reg_s2 = 0.f;     // owned channel = tid

  for (int ni = 0; ni < NPB; ++ni) {
    int node = blockIdx.x * NPB + ni;
    __syncthreads();
    fill_tile(x, eidx, eattr, node, tid, ldsA, ldsE);
    __syncthreads();

    v8f acc[4];
    node_matmul(ldsA, WhT, wr, wc, l15, half_sel, acc);

    int slot = wr * 2 + half_sel;       // 4 unique contributors per channel
#pragma unroll
    for (int t = 0; t < 4; ++t) {
      int cg = wc * 64 + t * 16 + l15;
      float bb = bias[cg], wl = wlast[cg];
      float sv1 = 0.f, sv2 = 0.f;
#pragma unroll
      for (int v = 0; v < 8; ++v) {
        int m = wr * 16 + v + 8 * half_sel;
        float g = acc[t][v] + bb + ldsE[m] * wl;
        sv1 += g; sv2 += g * g;
      }
      st1[slot][cg] = sv1;
      st2[slot][cg] = sv2;
    }
    __syncthreads();
    reg_s1 += st1[0][tid] + st1[1][tid] + st1[2][tid] + st1[3][tid];
    reg_s2 += st2[0][tid] + st2[1][tid] + st2[2][tid] + st2[3][tid];
  }
  partH[(blockIdx.x * 2 + 0) * HC + tid] = reg_s1;
  partH[(blockIdx.x * 2 + 1) * HC + tid] = reg_s2;
}

// finalize BN: bn_out[c] = gamma*rsqrt(var+eps), bn_out[C+c] = beta - mu*scale
__global__ void bnfin_kernel(const float* __restrict__ partial, int nblk, int C,
                             float invR, const float* __restrict__ gamma,
                             const float* __restrict__ beta,
                             float* __restrict__ bn_out) {
  __shared__ float r1[256], r2[256];
  int c = blockIdx.x, t = threadIdx.x;
  float s1 = 0.f, s2 = 0.f;
  for (int j = t; j < nblk; j += 256) {
    s1 += partial[(j * 2 + 0) * C + c];
    s2 += partial[(j * 2 + 1) * C + c];
  }
  r1[t] = s1; r2[t] = s2; __syncthreads();
  for (int ofs = 128; ofs > 0; ofs >>= 1) {
    if (t < ofs) { r1[t] += r1[t + ofs]; r2[t] += r2[t + ofs]; }
    __syncthreads();
  }
  if (t == 0) {
    float mu  = r1[0] * invR;
    float var = r2[0] * invR - mu * mu;
    float sc  = gamma[c] * rsqrtf(var + EPS);
    bn_out[c] = sc;
    bn_out[C + c] = beta[c] - mu * sc;
  }
}

// ---------------------------------------------------------------------------
// Pass 2: recompute g, apply hidden BN, gated sum over M -> summed[b,n,c],
// plus per-channel partial stats of `summed` for the output BN.
// ---------------------------------------------------------------------------
__global__ void __launch_bounds__(256)
main_kernel(const float* __restrict__ x, const int* __restrict__ eidx,
            const float* __restrict__ eattr, const h16* __restrict__ WhT,
            const float* __restrict__ bias, const float* __restrict__ wlast,
            const float* __restrict__ bnH, float* __restrict__ sumbuf,
            float* __restrict__ partO) {
  __shared__ h16   ldsA[32 * APAD];
  __shared__ float ldsE[M_SZ];
  __shared__ float gbuf[32 * GPAD];
  __shared__ float ssum[256];

  int tid = threadIdx.x;
  int lane = tid & 31, wid = tid >> 5;
  int wr = wid >> 2, wc = wid & 3;
  int half_sel = lane >> 4, l15 = lane & 15;
  float o1 = 0.f, o2 = 0.f;            // stats for channel tid (<128)

  for (int ni = 0; ni < NPB; ++ni) {
    int node = blockIdx.x * NPB + ni;
    __syncthreads();
    fill_tile(x, eidx, eattr, node, tid, ldsA, ldsE);
    __syncthreads();

    v8f acc[4];
    node_matmul(ldsA, WhT, wr, wc, l15, half_sel, acc);

#pragma unroll
    for (int t = 0; t < 4; ++t) {
      int cg = wc * 64 + t * 16 + l15;
      float bb = bias[cg], wl = wlast[cg];
      float sc = bnH[cg], sh = bnH[HC + cg];
#pragma unroll
      for (int v = 0; v < 8; ++v) {
        int m = wr * 16 + v + 8 * half_sel;
        float g = acc[t][v] + bb + ldsE[m] * wl;
        gbuf[m * GPAD + cg] = g * sc + sh;
      }
    }
    __syncthreads();

    // summed[c] = sum_m sigmoid(filt)*relu(core); filt=c, core=c+128
    {
      int c = tid & 127, mh = tid >> 7;
      float sp = 0.f;
      for (int mm = mh * 16; mm < mh * 16 + 16; ++mm) {
        float f  = gbuf[mm * GPAD + c];
        float co = gbuf[mm * GPAD + HA + c];
        sp += (1.f / (1.f + __expf(-f))) * fmaxf(co, 0.f);
      }
      ssum[tid] = sp;
    }
    __syncthreads();
    if (tid < HA) {
      float s = ssum[tid] + ssum[tid + 128];
      sumbuf[node * HA + tid] = s;
      o1 += s; o2 += s * s;
    }
  }
  if (tid < HA) {
    partO[(blockIdx.x * 2 + 0) * HA + tid] = o1;
    partO[(blockIdx.x * 2 + 1) * HA + tid] = o2;
  }
}

// residual + relu + time modulation, in place on x
__global__ void update_kernel(float* __restrict__ x, const float* __restrict__ sumbuf,
                              const float* __restrict__ bn, const float* __restrict__ tvec,
                              const float* __restrict__ twl, const float* __restrict__ tbl) {
  int idx = blockIdx.x * blockDim.x + threadIdx.x;   // B*N*HA
  int c  = idx & 127;
  int bi = idx >> 17;                                // / (N_SZ*HA)
  float s  = sumbuf[idx] * bn[c] + bn[HA + c];
  float xn = fmaxf(x[idx] + s, 0.f);
  float ts = tvec[bi];
  float sg = 1.f / (1.f + __expf(-ts * twl[c]));
  x[idx] = xn * sg + tanhf(ts * tbl[c]);
}

// final scalar: sum(x @ eo_w) + B*N*eo_b  (single block -> deterministic)
__global__ void out_kernel(const float* __restrict__ x, const float* __restrict__ eo_w,
                           const float* __restrict__ eo_b, float* __restrict__ out) {
  __shared__ float red[1024];
  int t = threadIdx.x;
  float s = 0.f;
  for (int i = t; i < B_SZ * N_SZ * HA; i += 1024) s += x[i] * eo_w[i & 127];
  red[t] = s; __syncthreads();
  for (int ofs = 512; ofs > 0; ofs >>= 1) {
    if (t < ofs) red[t] += red[t + ofs];
    __syncthreads();
  }
  if (t == 0) out[0] = red[0] + eo_b[0] * (float)(B_SZ * N_SZ);
}

// ---------------------------------------------------------------------------
extern "C" void kernel_launch(void* const* d_in, const int* in_sizes, int n_in,
                              void* d_out, int out_size, void* d_ws, size_t ws_size,
                              hipStream_t stream) {
  (void)in_sizes; (void)n_in; (void)out_size; (void)ws_size;

  const int*   node_attr = (const int*)  d_in[0];
  const float* edge_attr = (const float*)d_in[1];
  const int*   edge_idx  = (const int*)  d_in[2];
  const float* tvec      = (const float*)d_in[3];
  const float* emb       = (const float*)d_in[4];
  const float* W         = (const float*)d_in[5];
  const float* bvec      = (const float*)d_in[6];
  const float* g_h       = (const float*)d_in[7];
  const float* b_h       = (const float*)d_in[8];
  const float* g_o       = (const float*)d_in[9];
  const float* b_o       = (const float*)d_in[10];
  const float* tw        = (const float*)d_in[11];
  const float* tb        = (const float*)d_in[12];
  const float* eo_w      = (const float*)d_in[13];
  const float* eo_b      = (const float*)d_in[14];

  // workspace layout (all 256B-aligned), total ~14.9 MB
  char*  ws     = (char*)d_ws;
  float* x      = (float*)(ws + 0);                   // 4 MB  (B*N*HA f32)
  float* sumbuf = (float*)(ws + 4194304);             // 4 MB
  h16*   WhT    = (h16*)  (ws + 8388608);             // 128 KB (256x256 f16)
  float* partH  = (float*)(ws + 8519680);             // NBLK*2*256*4 = 4 MB
  float* partO  = (float*)(ws + 12713984);            // NBLK*2*128*4 = 2 MB
  float* bnH    = (float*)(ws + 14811136);            // 512 f32
  float* bnO    = (float*)(ws + 14813184);            // 256 f32

  embed_kernel<<<4096, 256, 0, stream>>>(node_attr, emb, x);

  for (int l = 0; l < 3; ++l) {
    const float* Wl    = W + (size_t)l * 257 * HC;
    const float* wlast = Wl + 256 * HC;               // edge-feature row (k=256)
    convw_kernel<<<256, 256, 0, stream>>>(Wl, WhT);
    stats_kernel<<<NBLK, 256, 0, stream>>>(x, edge_idx, edge_attr, WhT,
                                           bvec + l * HC, wlast, partH);
    bnfin_kernel<<<HC, 256, 0, stream>>>(partH, NBLK, HC,
                                         1.f / (float)(B_SZ * N_SZ * M_SZ),
                                         g_h + l * HC, b_h + l * HC, bnH);
    main_kernel<<<NBLK, 256, 0, stream>>>(x, edge_idx, edge_attr, WhT,
                                          bvec + l * HC, wlast, bnH, sumbuf, partO);
    bnfin_kernel<<<HA, 256, 0, stream>>>(partO, NBLK, HA,
                                         1.f / (float)(B_SZ * N_SZ),
                                         g_o + l * HA, b_o + l * HA, bnO);
    update_kernel<<<4096, 256, 0, stream>>>(x, sumbuf, bnO, tvec,
                                            tw + l * HA, tb + l * HA);
  }

  out_kernel<<<1, 1024, 0, stream>>>(x, eo_w, eo_b, (float*)d_out);
}
